// HyenaBlock_24610162606456
// MI455X (gfx1250) — matrix operand order repaired and checked by reading, not compiled, not run on hardware
//
#include <hip/hip_runtime.h>
#include <hip/hip_bf16.h>
#include <cstdint>

typedef __attribute__((ext_vector_type(16))) __bf16 v16bf;
typedef __attribute__((ext_vector_type(8)))  float    v8f;
typedef __attribute__((ext_vector_type(4)))  unsigned u32x4;
typedef __attribute__((ext_vector_type(8)))  unsigned u32x8;

static constexpr int Bn = 4;
static constexpr int Ln = 4096;
static constexpr int Dn = 1024;
static constexpr int Mn = Bn * Ln;   // 16384 GEMM rows

union Frag { uint4 q[2]; v16bf v; };   // 32B: one WMMA bf16 A/B fragment per lane

// --- CDNA5 async global->LDS staging (ASYNCcnt path) -----------------------
__device__ __forceinline__ unsigned lds_off(const void* p) {
  return (unsigned)(uintptr_t)p;       // addr[31:0] = wave-relative LDS offset
}
__device__ __forceinline__ void async_copy16(unsigned ldsoff, const void* g) {
  asm volatile("global_load_async_to_lds_b128 %0, %1, off"
               :: "v"(ldsoff), "v"(g) : "memory");
}
template <int N>
__device__ __forceinline__ void wait_async_le() {
#if __has_builtin(__builtin_amdgcn_s_wait_asynccnt)
  __builtin_amdgcn_s_wait_asynccnt(N);
#else
  asm volatile("s_wait_asynccnt %0" :: "i"(N) : "memory");
#endif
}
__device__ __forceinline__ void wait_ds0() {
  asm volatile("s_wait_dscnt 0x0" ::: "memory");
}

// --- CDNA5 Tensor Data Mover (TENSORcnt path) ------------------------------
// 1-D contiguous tile, 8-byte elements, D# per ISA 8.3/8.4 (count=1, type=2).
__device__ __forceinline__ void tdm_load_1d(unsigned lds_addr, const void* gptr,
                                            unsigned nbytes) {
  const unsigned long long ga = (unsigned long long)(uintptr_t)gptr;
  const unsigned ne = nbytes >> 3;                       // 8B elements (<=65535)
  u32x4 g0;
  g0[0] = 1u;                                            // count=1, user mode
  g0[1] = lds_addr;                                      // lds_addr[31:0]
  g0[2] = (unsigned)(ga & 0xFFFFFFFFu);                  // global_addr[31:0]
  g0[3] = (unsigned)((ga >> 32) & 0x01FFFFFFu) | (2u << 30);  // addr[56:32], type=2
  u32x8 g1;
  g1[0] = 3u << 16;                                      // mask=0, data_size=3 (8B)
  g1[1] = (ne & 0xFFFFu) << 16;                          // tensor_dim0[15:0] @bit48
  g1[2] = ((ne >> 16) & 0xFFFFu) | (1u << 16);           // tensor_dim0[31:16], dim1=1
  g1[3] = (ne & 0xFFFFu) << 16;                          // tile_dim0 @bit112
  g1[4] = 0u;                                            // tile_dim1/2 unused
  g1[5] = ne;                                            // tensor_dim0_stride[31:0]
  g1[6] = 0u;
  g1[7] = 0u;
  asm volatile("tensor_load_to_lds %0, %1" :: "s"(g0), "s"(g1) : "memory");
}
__device__ __forceinline__ void wait_tensor0() {
#if __has_builtin(__builtin_amdgcn_s_wait_tensorcnt)
  __builtin_amdgcn_s_wait_tensorcnt(0);
#else
  asm volatile("s_wait_tensorcnt 0x0" ::: "memory");
#endif
}

// ---------------------------------------------------------------------------
// Kernel 1: bf16 copies. xbf = bf16(x); W*T = bf16 transpose [N,K];
// hT[d][l] = bf16(h_filter[l,d] * exp(-0.01 l))
// ---------------------------------------------------------------------------
__global__ __launch_bounds__(256) void prep_kernel(
    const float* __restrict__ x,
    const float* __restrict__ Wu, const float* __restrict__ Wv,
    const float* __restrict__ Wo, const float* __restrict__ hf,
    __bf16* __restrict__ xbf,
    __bf16* __restrict__ WuT, __bf16* __restrict__ WvT,
    __bf16* __restrict__ WoT, __bf16* __restrict__ hT) {
  int idx = blockIdx.x * blockDim.x + threadIdx.x;
  if (idx < Mn * Dn) xbf[idx] = (__bf16)x[idx];
  if (idx < Dn * Dn) {
    int k = idx / Dn, n = idx % Dn;
    WuT[(size_t)n * Dn + k] = (__bf16)Wu[idx];
    WvT[(size_t)n * Dn + k] = (__bf16)Wv[idx];
    WoT[(size_t)n * Dn + k] = (__bf16)Wo[idx];
  }
  if (idx < Ln * Dn) {
    int l = idx / Dn, d = idx % Dn;
    hT[(size_t)d * Ln + l] = (__bf16)(hf[idx] * __expf(-0.01f * (float)l));
  }
}

// ---------------------------------------------------------------------------
// Fragment layouts (ISA 7.12.2, wave32, 16x16x32 bf16):
//  A: lane m=lane&15, g=lane>>4; elems 0..7 -> K=g*8+e ; 8..15 -> K=16+g*8+(e-8)
//  B: lane n=lane&15, g=lane>>4; elems 0..7 -> K=g*8+e ; 8..15 -> K=16+g*8+(e-8)
//  C/D f32: lane n=lane&15; VGPR r -> M=r+8g
// ---------------------------------------------------------------------------
__device__ __forceinline__ Frag load_a_frag(const __bf16* As, int row, int g) {
  Frag f;
  f.q[0] = *(const uint4*)&As[row * 32 + g * 8];
  f.q[1] = *(const uint4*)&As[row * 32 + 16 + g * 8];
  return f;
}
__device__ __forceinline__ Frag load_b_frag(const __bf16* Bs, int n, int g) {
  Frag f;
  f.q[0] = *(const uint4*)&Bs[n * 32 + g * 8];
  f.q[1] = *(const uint4*)&Bs[n * 32 + 16 + g * 8];
  return f;
}

// ---------------------------------------------------------------------------
// Kernel 2: u = x@Wu + bu ; v = x@Wv + bv. Block tile 128x64, BK=32,
// double-buffered async LDS staging, 8 waves (4x2), 2x2 WMMA subtiles/wave.
// ---------------------------------------------------------------------------
__global__ __launch_bounds__(256) void gemm_uv_kernel(
    const __bf16* __restrict__ xbf,
    const __bf16* __restrict__ WuT, const __bf16* __restrict__ WvT,
    const float* __restrict__ bu, const float* __restrict__ bv,
    float* __restrict__ u, __bf16* __restrict__ vT) {
  __shared__ __bf16 As[2][128 * 32];   // 2 x 8 KB
  __shared__ __bf16 Bus[2][64 * 32];   // 2 x 4 KB
  __shared__ __bf16 Bvs[2][64 * 32];   // 2 x 4 KB
  const int tid = threadIdx.x;
  const int bm = blockIdx.x >> 4, bn = blockIdx.x & 15;
  const int row0 = bm * 128, col0 = bn * 64;
  const int wave = tid >> 5, lane = tid & 31;
  const int wm = wave & 3, wn = wave >> 2;
  const int g = lane >> 4, mn = lane & 15;
  const int crow = tid >> 2, cseg = (tid & 3) * 8;

  auto stage = [&](int buf, int k0) {   // 4 async b128 per wave per stage
    async_copy16(lds_off(&As[buf][crow * 32 + cseg]),
                 xbf + (size_t)(row0 + crow) * Dn + k0 + cseg);
    async_copy16(lds_off(&As[buf][(crow + 64) * 32 + cseg]),
                 xbf + (size_t)(row0 + crow + 64) * Dn + k0 + cseg);
    async_copy16(lds_off(&Bus[buf][crow * 32 + cseg]),
                 WuT + (size_t)(col0 + crow) * Dn + k0 + cseg);
    async_copy16(lds_off(&Bvs[buf][crow * 32 + cseg]),
                 WvT + (size_t)(col0 + crow) * Dn + k0 + cseg);
  };

  v8f cu[2][2] = {{{}, {}}, {{}, {}}};
  v8f cv[2][2] = {{{}, {}}, {{}, {}}};

  constexpr int NSTEP = Dn / 32;       // 32
  stage(0, 0);
  for (int i = 0; i < NSTEP; ++i) {
    const int cur = i & 1;
    if (i + 1 < NSTEP) {               // prefetch next tile into other buffer
      stage(cur ^ 1, (i + 1) * 32);
      wait_async_le<4>();              // tile i complete (in-order returns)
    } else {
      wait_async_le<0>();
    }
    __syncthreads();

    Frag a[2], fu[2], fv[2];
    #pragma unroll
    for (int s = 0; s < 2; ++s) {
      a[s]  = load_a_frag(As[cur],  wm * 32 + s * 16 + mn, g);
      fu[s] = load_b_frag(Bus[cur], wn * 32 + s * 16 + mn, g);
      fv[s] = load_b_frag(Bvs[cur], wn * 32 + s * 16 + mn, g);
    }
    #pragma unroll
    for (int sm = 0; sm < 2; ++sm)
      #pragma unroll
      for (int sn = 0; sn < 2; ++sn) {
        cu[sm][sn] = __builtin_amdgcn_wmma_f32_16x16x32_bf16(
            false, a[sm].v, false, fu[sn].v, (short)0, cu[sm][sn], false, false);
        cv[sm][sn] = __builtin_amdgcn_wmma_f32_16x16x32_bf16(
            false, a[sm].v, false, fv[sn].v, (short)0, cv[sm][sn], false, false);
      }
    wait_ds0();                        // fragment reads retired before refill
    __syncthreads();
  }

  const int batch = row0 / Ln;
  #pragma unroll
  for (int sm = 0; sm < 2; ++sm) {
    #pragma unroll
    for (int sn = 0; sn < 2; ++sn) {
      const int col = col0 + wn * 32 + sn * 16 + mn;
      const float biasu = bu[col], biasv = bv[col];
      #pragma unroll
      for (int r = 0; r < 8; ++r) {
        const int row = row0 + wm * 32 + sm * 16 + r + 8 * g;
        u[(size_t)row * Dn + col] = cu[sm][sn][r] + biasu;
        vT[((size_t)col * Ln + (row - batch * Ln)) * 4 + batch] =
            (__bf16)(cv[sm][sn][r] + biasv);
      }
    }
  }
}

// ---------------------------------------------------------------------------
// Kernel 3: causal Toeplitz conv per channel via WMMA + gating multiply.
// One workgroup per channel d; h_d (8KB) and v_d (32KB) staged by the
// Tensor Data Mover (TENSORcnt), issued once by wave 0.
// ---------------------------------------------------------------------------
__global__ __launch_bounds__(256) void conv_mul_kernel(
    const __bf16* __restrict__ hT, const __bf16* __restrict__ vT,
    const float* __restrict__ u, __bf16* __restrict__ ymulbf) {
  __shared__ __bf16 hS[Ln];          // 8 KB
  __shared__ __bf16 vS[Ln * 4];      // 32 KB  (vS[t*4 + b])
  const int d = blockIdx.x;
  const int tid = threadIdx.x;
  const int wave = tid >> 5, lane = tid & 31;

  if (wave == 0) {                   // one TDM descriptor per tile
    tdm_load_1d(lds_off(&hS[0]), hT + (size_t)d * Ln,     Ln * 2);
    tdm_load_1d(lds_off(&vS[0]), vT + (size_t)d * Ln * 4, Ln * 4 * 2);
    wait_tensor0();
  }
  __syncthreads();

  const int g = lane >> 4, mn = lane & 15;

  for (int rt = wave; rt < Ln / 16; rt += 8) {
    const int l0 = rt * 16;
    const int lrow = l0 + mn;        // A-fragment row (output time index)
    v8f c = {};
    for (int t0 = 0; t0 <= l0 + 15; t0 += 32) {
      v16bf a, b;
      #pragma unroll
      for (int e = 0; e < 16; ++e) {
        const int kk = (e < 8) ? (g * 8 + e) : (g * 8 + e + 8);
        const int delta = lrow - (t0 + kk);               // causal mask
        a[e] = (delta >= 0) ? hS[delta] : (__bf16)0.0f;
      }
      #pragma unroll
      for (int e = 0; e < 16; ++e) {
        const int t = t0 + ((e < 8) ? (g * 8 + e) : (g * 8 + e + 8));
        b[e] = (mn < 4 && t < Ln) ? vS[t * 4 + mn] : (__bf16)0.0f;
      }
      c = __builtin_amdgcn_wmma_f32_16x16x32_bf16(false, a, false, b, (short)0, c, false, false);
    }
    if (mn < 4) {                    // only 4 of 16 N-columns are real batches
      #pragma unroll
      for (int r = 0; r < 8; ++r) {
        const int l = l0 + r + 8 * g;
        const size_t off = ((size_t)mn * Ln + l) * Dn + d;
        ymulbf[off] = (__bf16)(c[r] * u[off]);
      }
    }
  }
}

// ---------------------------------------------------------------------------
// Kernel 4: out = ymul @ Wo + bo (double-buffered LDS-tiled WMMA GEMM)
// ---------------------------------------------------------------------------
__global__ __launch_bounds__(256) void gemm_o_kernel(
    const __bf16* __restrict__ ym, const __bf16* __restrict__ WoT,
    const float* __restrict__ bo, float* __restrict__ out) {
  __shared__ __bf16 As[2][128 * 32];   // 2 x 8 KB
  __shared__ __bf16 Bs[2][64 * 32];    // 2 x 4 KB
  const int tid = threadIdx.x;
  const int bm = blockIdx.x >> 4, bn = blockIdx.x & 15;
  const int row0 = bm * 128, col0 = bn * 64;
  const int wave = tid >> 5, lane = tid & 31;
  const int wm = wave & 3, wn = wave >> 2;
  const int g = lane >> 4, mn = lane & 15;
  const int crow = tid >> 2, cseg = (tid & 3) * 8;

  auto stage = [&](int buf, int k0) {   // 3 async b128 per wave per stage
    async_copy16(lds_off(&As[buf][crow * 32 + cseg]),
                 ym + (size_t)(row0 + crow) * Dn + k0 + cseg);
    async_copy16(lds_off(&As[buf][(crow + 64) * 32 + cseg]),
                 ym + (size_t)(row0 + crow + 64) * Dn + k0 + cseg);
    async_copy16(lds_off(&Bs[buf][crow * 32 + cseg]),
                 WoT + (size_t)(col0 + crow) * Dn + k0 + cseg);
  };

  v8f c[2][2] = {{{}, {}}, {{}, {}}};

  constexpr int NSTEP = Dn / 32;
  stage(0, 0);
  for (int i = 0; i < NSTEP; ++i) {
    const int cur = i & 1;
    if (i + 1 < NSTEP) {
      stage(cur ^ 1, (i + 1) * 32);
      wait_async_le<3>();
    } else {
      wait_async_le<0>();
    }
    __syncthreads();

    Frag a[2], fb[2];
    #pragma unroll
    for (int s = 0; s < 2; ++s) {
      a[s]  = load_a_frag(As[cur], wm * 32 + s * 16 + mn, g);
      fb[s] = load_b_frag(Bs[cur], wn * 32 + s * 16 + mn, g);
    }
    #pragma unroll
    for (int sm = 0; sm < 2; ++sm)
      #pragma unroll
      for (int sn = 0; sn < 2; ++sn)
        c[sm][sn] = __builtin_amdgcn_wmma_f32_16x16x32_bf16(
            false, a[sm].v, false, fb[sn].v, (short)0, c[sm][sn], false, false);
    wait_ds0();
    __syncthreads();
  }

  #pragma unroll
  for (int sm = 0; sm < 2; ++sm) {
    #pragma unroll
    for (int sn = 0; sn < 2; ++sn) {
      const int col = col0 + wn * 32 + sn * 16 + mn;
      const float bias = bo[col];
      #pragma unroll
      for (int r = 0; r < 8; ++r) {
        const int row = row0 + wm * 32 + sm * 16 + r + 8 * g;
        out[(size_t)row * Dn + col] = c[sm][sn][r] + bias;
      }
    }
  }
}

// ---------------------------------------------------------------------------
extern "C" void kernel_launch(void* const* d_in, const int* in_sizes, int n_in,
                              void* d_out, int out_size, void* d_ws, size_t ws_size,
                              hipStream_t stream) {
  const float* x  = (const float*)d_in[0];
  const float* Wu = (const float*)d_in[1];
  const float* bu = (const float*)d_in[2];
  const float* Wv = (const float*)d_in[3];
  const float* bv = (const float*)d_in[4];
  const float* hf = (const float*)d_in[5];
  const float* Wo = (const float*)d_in[6];
  const float* bo = (const float*)d_in[7];
  float* out = (float*)d_out;

  char* ws = (char*)d_ws;
  size_t off = 0;
  auto alloc = [&](size_t bytes) -> void* {
    off = (off + 255) & ~(size_t)255;
    void* p = ws + off;
    off += bytes;
    return p;
  };
  __bf16* xbf    = (__bf16*)alloc((size_t)Mn * Dn * 2);       // 32 MB
  __bf16* WuT    = (__bf16*)alloc((size_t)Dn * Dn * 2);
  __bf16* WvT    = (__bf16*)alloc((size_t)Dn * Dn * 2);
  __bf16* WoT    = (__bf16*)alloc((size_t)Dn * Dn * 2);
  __bf16* hT     = (__bf16*)alloc((size_t)Dn * Ln * 2);       // 8 MB
  float*  u      = (float*) alloc((size_t)Mn * Dn * 4);       // 64 MB
  __bf16* vT     = (__bf16*)alloc((size_t)Dn * Ln * 4 * 2);   // 32 MB
  __bf16* ymulbf = (__bf16*)alloc((size_t)Mn * Dn * 2);       // 32 MB

  const int gemm_blocks = (Mn / 128) * (Dn / 64);             // 2048

  prep_kernel<<<(Mn * Dn) / 256, 256, 0, stream>>>(x, Wu, Wv, Wo, hf,
                                                   xbf, WuT, WvT, WoT, hT);
  gemm_uv_kernel<<<gemm_blocks, 256, 0, stream>>>(xbf, WuT, WvT, bu, bv, u, vT);
  conv_mul_kernel<<<Dn, 256, 0, stream>>>(hT, vT, u, ymulbf);
  gemm_o_kernel<<<gemm_blocks, 256, 0, stream>>>(ymulbf, WoT, bo, out);
}
